// GroupedQueryAttention_70815420776976
// MI455X (gfx1250) — compile-verified, matrix operand order checked by reading, not run on hardware
//
#include <hip/hip_runtime.h>

typedef _Float16 half_t;
typedef __attribute__((ext_vector_type(16))) _Float16 v16h;
typedef __attribute__((ext_vector_type(8)))  _Float16 v8h;
typedef __attribute__((ext_vector_type(8)))  float    v8f;

#define B_    2
#define N_    4096
#define D_    1024
#define HQ_   16
#define HKV_  4
#define HD_   64
#define KVD_  256            // HKV*HD
#define MROWS 8192           // B*N

#ifndef __has_builtin
#define __has_builtin(x) 0
#endif
#if __has_builtin(__builtin_amdgcn_tensor_load_to_lds) && \
    __has_builtin(__builtin_amdgcn_s_wait_tensorcnt)
#define HAVE_TDM 1
#else
#define HAVE_TDM 0
#endif

// ---------------------------------------------------------------------------
// helpers
// ---------------------------------------------------------------------------
__device__ __forceinline__ v16h pack16(v8h lo, v8h hi) {
  v16h r;
#pragma unroll
  for (int j = 0; j < 8; ++j) { r[j] = lo[j]; r[j + 8] = hi[j]; }
  return r;
}

#if HAVE_TDM
typedef unsigned int u32x4 __attribute__((ext_vector_type(4)));
typedef int          i32x4 __attribute__((ext_vector_type(4)));
typedef int          i32x8 __attribute__((ext_vector_type(8)));

// Issue a 2D TDM load: tile_h rows x tile_w f16 elements, row stride
// row_stride_elems (f16 units), global -> LDS at byte offset lds_off.
// D# per CDNA5 ISA 8.3/8.4: group0 {count=1, lds_addr, global_addr, type=2},
// group1 {data_size=1(2B), tensor_dim0/1, tile_dim0/1, tensor_dim0_stride}.
__device__ __forceinline__ void tdm_load_2d(unsigned lds_off, const void* gaddr,
                                            unsigned tile_w, unsigned tile_h,
                                            unsigned row_stride_elems) {
  unsigned long long q0 = 1ull | ((unsigned long long)lds_off << 32);
  unsigned long long ga = (unsigned long long)(uintptr_t)gaddr;
  unsigned long long q1 = (ga & 0x01FFFFFFFFFFFFFFull) | (2ull << 62);
  u32x4 g0;
  g0.x = (unsigned)q0; g0.y = (unsigned)(q0 >> 32);
  g0.z = (unsigned)q1; g0.w = (unsigned)(q1 >> 32);

  unsigned long long p0 = (1ull << 16)                                  // data_size = 2B
                        | (((unsigned long long)tile_w & 0xFFFF) << 48); // tensor_dim0 lo16
  unsigned long long p1 = (((unsigned long long)tile_w >> 16) & 0xFFFF)  // tensor_dim0 hi16
                        | ((unsigned long long)tile_h << 16)             // tensor_dim1
                        | (((unsigned long long)tile_w & 0xFFFF) << 48); // tile_dim0
  unsigned long long p2 = ((unsigned long long)tile_h & 0xFFFF)          // tile_dim1
                        | ((unsigned long long)row_stride_elems << 32);  // dim0_stride lo32
  unsigned long long p3 = 0;                                             // stride hi / dim1_stride
  i32x8 g1;
  g1[0] = (int)p0; g1[1] = (int)(p0 >> 32);
  g1[2] = (int)p1; g1[3] = (int)(p1 >> 32);
  g1[4] = (int)p2; g1[5] = (int)(p2 >> 32);
  g1[6] = (int)p3; g1[7] = (int)(p3 >> 32);
  i32x4 z4 = {0, 0, 0, 0};
#if defined(__clang_major__) && (__clang_major__ >= 23)
  i32x8 z8 = {0, 0, 0, 0, 0, 0, 0, 0};
  __builtin_amdgcn_tensor_load_to_lds(g0, g1, z4, z4, z8, 0);
#else
  __builtin_amdgcn_tensor_load_to_lds(g0, g1, z4, z4, 0);
#endif
}
#endif  // HAVE_TDM

// ---------------------------------------------------------------------------
// f32 -> f16 conversion
// ---------------------------------------------------------------------------
__global__ void cvt_f32_f16(const float* __restrict__ src, half_t* __restrict__ dst, int n) {
  int i = blockIdx.x * blockDim.x + threadIdx.x;
  if (i < n) dst[i] = (half_t)src[i];
}

// W (K x Nc, f32, row-major) -> Wt (Nc x K, f16): WMMA B-fragments contiguous
__global__ void wtrans_f32_f16(const float* __restrict__ W, half_t* __restrict__ Wt,
                               int K, int Nc) {
  int i = blockIdx.x * blockDim.x + threadIdx.x;
  if (i >= K * Nc) return;
  int n = i / K, k = i - n * K;
  Wt[(size_t)n * K + k] = (half_t)W[(size_t)k * Nc + n];
}

// Vh [b][n][hkv][hd] -> Vt [b][hkv][hd][n]
__global__ void vtrans_f16(const half_t* __restrict__ Vh, half_t* __restrict__ Vt) {
  int i = blockIdx.x * blockDim.x + threadIdx.x;
  if (i >= B_ * KVD_ * N_) return;
  int n = i & (N_ - 1);
  int rest = i >> 12;
  int d = rest & (HD_ - 1); rest >>= 6;
  int h = rest & (HKV_ - 1);
  int b = rest >> 2;
  Vt[i] = Vh[(((size_t)(b * N_ + n)) * HKV_ + h) * HD_ + d];
}

// ---------------------------------------------------------------------------
// in-place RoPE on [b][n][nh][64] f16; thread handles pair (d, d+32)
// ---------------------------------------------------------------------------
__global__ void rope_f16(half_t* __restrict__ X, int nheads, int total) {
  int i = blockIdx.x * blockDim.x + threadIdx.x;
  if (i >= total) return;
  int d = i & 31;
  int rest = i >> 5;                          // (b*N + n)*nheads + h
  int pos = (rest / nheads) & (N_ - 1);
  size_t base = (size_t)rest * HD_;
  float invf = __expf(-__logf(10000.0f) * ((float)d * (1.0f / 32.0f)));
  float th = (float)pos * invf;
  float s, c;
  __sincosf(th, &s, &c);
  float x0 = (float)X[base + d];
  float x1 = (float)X[base + d + 32];
  X[base + d]      = (half_t)(x0 * c - x1 * s);
  X[base + d + 32] = (half_t)(x1 * c + x0 * s);
}

// ---------------------------------------------------------------------------
// WMMA GEMM:  C[M x Nc] = A[M x K] * Bt^T,  Bt given as [Nc x K] (f16).
// Wave computes 32x64 of C (2 A-frags x 4 B-frags -> 8 WMMA per k-step);
// block = 2x2 waves -> 64 x 128 C tile.
// ---------------------------------------------------------------------------
template <bool OUT_F32>
__global__ __launch_bounds__(128)
void gemm_wmma(const half_t* __restrict__ A, const half_t* __restrict__ Bt,
               void* __restrict__ Cout, int Nc, int K) {
  const int lane = threadIdx.x & 31;
  const int wave = threadIdx.x >> 5;
  const int wm = wave >> 1, wn = wave & 1;
  const int r  = lane & 15;
  const int hi = lane >> 4;
  const int tileM = blockIdx.y * 64 + wm * 32;
  const int tileN = blockIdx.x * 128 + wn * 64;

  const half_t* arow0 = A + (size_t)(tileM + r) * K;
  const half_t* arow1 = A + (size_t)(tileM + 16 + r) * K;
  const half_t* brow[4] = { Bt + (size_t)(tileN + r) * K,
                            Bt + (size_t)(tileN + 16 + r) * K,
                            Bt + (size_t)(tileN + 32 + r) * K,
                            Bt + (size_t)(tileN + 48 + r) * K };

  v8f acc[2][4] = {};
  for (int k = 0; k < K; k += 32) {
    __builtin_prefetch((const void*)(arow0 + k + 512), 0, 1);  // global_prefetch_b8
    __builtin_prefetch((const void*)(brow[0] + k + 512), 0, 1);
    v16h a0 = pack16(*(const v8h*)(arow0 + k + hi * 8),
                     *(const v8h*)(arow0 + k + 16 + hi * 8));
    v16h a1 = pack16(*(const v8h*)(arow1 + k + hi * 8),
                     *(const v8h*)(arow1 + k + 16 + hi * 8));
#pragma unroll
    for (int n = 0; n < 4; ++n) {
      v16h bf = pack16(*(const v8h*)(brow[n] + k + hi * 8),
                       *(const v8h*)(brow[n] + k + 16 + hi * 8));
      acc[0][n] = __builtin_amdgcn_wmma_f32_16x16x32_f16(false, a0, false, bf,
                                                         (short)0, acc[0][n], false, false);
      acc[1][n] = __builtin_amdgcn_wmma_f32_16x16x32_f16(false, a1, false, bf,
                                                         (short)0, acc[1][n], false, false);
    }
  }
#pragma unroll
  for (int i = 0; i < 2; ++i)
#pragma unroll
    for (int n = 0; n < 4; ++n)
#pragma unroll
      for (int v = 0; v < 8; ++v) {
        size_t m  = (size_t)tileM + i * 16 + v + 8 * hi;
        size_t nn = (size_t)tileN + n * 16 + r;
        if (OUT_F32) ((float*)Cout)[m * Nc + nn] = acc[i][n][v];
        else         ((half_t*)Cout)[m * Nc + nn] = (half_t)acc[i][n][v];
      }
}

// ---------------------------------------------------------------------------
// Flash attention (non-causal GQA). One wave = 16 queries of one (b,h).
// 64-key tiles: K staged to LDS via double-buffered TDM (tensor_load_to_lds,
// s_wait_tensorcnt), 8 WMMA for S + 8 WMMA for P@V per tile. Online softmax
// with deferred (per-lane partial) row sums; 1/sqrt(HD) folded into Q.
// ---------------------------------------------------------------------------
__global__ __launch_bounds__(128)
void fa_kernel(const half_t* __restrict__ Qh, const half_t* __restrict__ Kh,
               const half_t* __restrict__ Vt, half_t* __restrict__ AOh) {
  __shared__ __align__(16) half_t kbuf[4][2][64][64];   // 64 KB: K double buffer
  __shared__ __align__(16) half_t pbuf[4][16][64];      // 8 KB: P staging
  const int lane = threadIdx.x & 31;
  const int wave = threadIdx.x >> 5;
  const int r  = lane & 15;
  const int hi = lane >> 4;

  const int w  = blockIdx.x * 4 + wave;       // 0 .. B*HQ*(N/16)-1
  const int qt = w & (N_ / 16 - 1);
  const int h  = (w >> 8) & (HQ_ - 1);
  const int b  = w >> 12;
  const int hk = h >> 2;                      // G = 4

  // Q fragments, with softmax scale folded in (0.125 exact in f16)
  const half_t* qrow = Qh + (((size_t)(b * N_ + qt * 16 + r)) * HQ_ + h) * HD_;
  v16h qa0 = pack16(*(const v8h*)(qrow + hi * 8),      *(const v8h*)(qrow + 16 + hi * 8));
  v16h qa1 = pack16(*(const v8h*)(qrow + 32 + hi * 8), *(const v8h*)(qrow + 48 + hi * 8));
#pragma unroll
  for (int j = 0; j < 16; ++j) { qa0[j] *= (half_t)0.125f; qa1[j] *= (half_t)0.125f; }

  v8f o[4]; o[0] = v8f{}; o[1] = v8f{}; o[2] = v8f{}; o[3] = v8f{};
  float mrow[8], lrow[8];
#pragma unroll
  for (int v = 0; v < 8; ++v) { mrow[v] = -1e30f; lrow[v] = 0.0f; }

  const half_t* kbase = Kh + ((size_t)b * N_ * HKV_ + hk) * HD_;     // + kpos*KVD
  const half_t* vbase = Vt + ((size_t)(b * HKV_ + hk)) * HD_ * N_;   // + d*N + n

  const unsigned lds_k0 = (unsigned)(unsigned long long)(uintptr_t)&kbuf[wave][0][0][0];
  const unsigned lds_k1 = (unsigned)(unsigned long long)(uintptr_t)&kbuf[wave][1][0][0];

  const int ITER = N_ / 64;
#if HAVE_TDM
  tdm_load_2d(lds_k0, (const void*)kbase, HD_, 64, KVD_);
#else
  for (int rr = lane; rr < 64; rr += 32)
#pragma unroll
    for (int cc = 0; cc < 64; cc += 8)
      *(v8h*)&kbuf[wave][0][rr][cc] = *(const v8h*)(kbase + (size_t)rr * KVD_ + cc);
#endif

  int buf = 0;
  for (int it = 0; it < ITER; ++it) {
    const int kv0 = it * 64;
#if HAVE_TDM
    if (it + 1 < ITER) {
      tdm_load_2d(buf ? lds_k0 : lds_k1,
                  (const void*)(kbase + (size_t)(kv0 + 64) * KVD_), HD_, 64, KVD_);
      __builtin_amdgcn_s_wait_tensorcnt((short)1);   // current tile complete
    } else {
      __builtin_amdgcn_s_wait_tensorcnt((short)0);
    }
    asm volatile("" ::: "memory");   // keep LDS reads below the tensorcnt wait
#else
    if (it + 1 < ITER) {
      const half_t* nsrc = kbase + (size_t)(kv0 + 64) * KVD_;
      half_t (*ndst)[64] = kbuf[wave][buf ^ 1];
      for (int rr = lane; rr < 64; rr += 32)
#pragma unroll
        for (int cc = 0; cc < 64; cc += 8)
          *(v8h*)&ndst[rr][cc] = *(const v8h*)(nsrc + (size_t)rr * KVD_ + cc);
    }
#endif

    // ---- S = (Q/8) K^T for 64 keys: 4 sub-tiles x 2 WMMA ----
    v8f s[4]; s[0] = v8f{}; s[1] = v8f{}; s[2] = v8f{}; s[3] = v8f{};
#pragma unroll
    for (int c = 0; c < 4; ++c) {
      const half_t* kr = &kbuf[wave][buf][c * 16 + r][0];
      v16h b0 = pack16(*(const v8h*)(kr + hi * 8),      *(const v8h*)(kr + 16 + hi * 8));
      v16h b1 = pack16(*(const v8h*)(kr + 32 + hi * 8), *(const v8h*)(kr + 48 + hi * 8));
      s[c] = __builtin_amdgcn_wmma_f32_16x16x32_f16(false, qa0, false, b0, (short)0, s[c], false, false);
      s[c] = __builtin_amdgcn_wmma_f32_16x16x32_f16(false, qa1, false, b1, (short)0, s[c], false, false);
    }

    // ---- online softmax (row max reduced over 16-lane half; sums deferred) ----
#pragma unroll
    for (int v = 0; v < 8; ++v) {
      float t0 = s[0][v], t1 = s[1][v], t2 = s[2][v], t3 = s[3][v];
      float mx = fmaxf(fmaxf(t0, t1), fmaxf(t2, t3));
      mx = fmaxf(mx, __shfl_xor(mx, 1));
      mx = fmaxf(mx, __shfl_xor(mx, 2));
      mx = fmaxf(mx, __shfl_xor(mx, 4));
      mx = fmaxf(mx, __shfl_xor(mx, 8));
      float mnew  = fmaxf(mrow[v], mx);
      float alpha = __expf(mrow[v] - mnew);
      float p0 = __expf(t0 - mnew), p1 = __expf(t1 - mnew);
      float p2 = __expf(t2 - mnew), p3 = __expf(t3 - mnew);
      lrow[v] = lrow[v] * alpha + (p0 + p1 + p2 + p3);   // per-lane partial sum
      mrow[v] = mnew;
      o[0][v] *= alpha; o[1][v] *= alpha; o[2][v] *= alpha; o[3][v] *= alpha;
      pbuf[wave][v + 8 * hi][r]      = (half_t)p0;
      pbuf[wave][v + 8 * hi][16 + r] = (half_t)p1;
      pbuf[wave][v + 8 * hi][32 + r] = (half_t)p2;
      pbuf[wave][v + 8 * hi][48 + r] = (half_t)p3;
    }

    // ---- reload P as two 16x32 A-fragments ----
    v16h pa0 = pack16(*(const v8h*)&pbuf[wave][r][hi * 8],
                      *(const v8h*)&pbuf[wave][r][16 + hi * 8]);
    v16h pa1 = pack16(*(const v8h*)&pbuf[wave][r][32 + hi * 8],
                      *(const v8h*)&pbuf[wave][r][48 + hi * 8]);

    // ---- O += P * V ----
#pragma unroll
    for (int t = 0; t < 4; ++t) {
      const half_t* vrow = vbase + (size_t)(t * 16 + r) * N_ + kv0;  // Vt[d][kv]
      v16h vb0 = pack16(*(const v8h*)(vrow + hi * 8),      *(const v8h*)(vrow + 16 + hi * 8));
      v16h vb1 = pack16(*(const v8h*)(vrow + 32 + hi * 8), *(const v8h*)(vrow + 48 + hi * 8));
      o[t] = __builtin_amdgcn_wmma_f32_16x16x32_f16(false, pa0, false, vb0, (short)0, o[t], false, false);
      o[t] = __builtin_amdgcn_wmma_f32_16x16x32_f16(false, pa1, false, vb1, (short)0, o[t], false, false);
    }
    buf ^= 1;
  }

  // ---- finish deferred row sums, normalize, store [b][n][hq][hd] ----
#pragma unroll
  for (int v = 0; v < 8; ++v) {
    float ls = lrow[v];
    ls += __shfl_xor(ls, 1);
    ls += __shfl_xor(ls, 2);
    ls += __shfl_xor(ls, 4);
    ls += __shfl_xor(ls, 8);
    float inv = 1.0f / ls;
    size_t orow = (((size_t)(b * N_ + qt * 16 + v + 8 * hi)) * HQ_ + h) * HD_;
#pragma unroll
    for (int t = 0; t < 4; ++t)
      AOh[orow + t * 16 + r] = (half_t)(o[t][v] * inv);
  }
}

// ---------------------------------------------------------------------------
// launch
// ---------------------------------------------------------------------------
extern "C" void kernel_launch(void* const* d_in, const int* in_sizes, int n_in,
                              void* d_out, int out_size, void* d_ws, size_t ws_size,
                              hipStream_t stream) {
  const float* x  = (const float*)d_in[0];
  const float* Wq = (const float*)d_in[1];
  const float* Wk = (const float*)d_in[2];
  const float* Wv = (const float*)d_in[3];
  const float* Wo = (const float*)d_in[4];
  float* out = (float*)d_out;

  half_t* ws = (half_t*)d_ws;
  size_t off = 0;
  half_t* xh  = ws + off; off += (size_t)MROWS * D_;
  half_t* Wqt = ws + off; off += (size_t)D_ * D_;
  half_t* Wkt = ws + off; off += (size_t)D_ * KVD_;
  half_t* Wvt = ws + off; off += (size_t)D_ * KVD_;
  half_t* Wot = ws + off; off += (size_t)D_ * D_;
  half_t* Qh  = ws + off; off += (size_t)MROWS * D_;
  half_t* Kh  = ws + off; off += (size_t)MROWS * KVD_;
  half_t* Vh  = ws + off; off += (size_t)MROWS * KVD_;
  half_t* Vt  = ws + off; off += (size_t)MROWS * KVD_;
  half_t* AOh = xh;   // x no longer needed after QKV projections

  const int TPB = 256;
  cvt_f32_f16<<<(MROWS * D_ + TPB - 1) / TPB, TPB, 0, stream>>>(x, xh, MROWS * D_);
  wtrans_f32_f16<<<(D_ * D_   + TPB - 1) / TPB, TPB, 0, stream>>>(Wq, Wqt, D_, D_);
  wtrans_f32_f16<<<(D_ * KVD_ + TPB - 1) / TPB, TPB, 0, stream>>>(Wk, Wkt, D_, KVD_);
  wtrans_f32_f16<<<(D_ * KVD_ + TPB - 1) / TPB, TPB, 0, stream>>>(Wv, Wvt, D_, KVD_);
  wtrans_f32_f16<<<(D_ * D_   + TPB - 1) / TPB, TPB, 0, stream>>>(Wo, Wot, D_, D_);

  // QKV projections (WMMA f16): block tile 64M x 128N
  gemm_wmma<false><<<dim3(D_ / 128,   MROWS / 64), 128, 0, stream>>>(xh, Wqt, (void*)Qh, D_,   D_);
  gemm_wmma<false><<<dim3(KVD_ / 128, MROWS / 64), 128, 0, stream>>>(xh, Wkt, (void*)Kh, KVD_, D_);
  gemm_wmma<false><<<dim3(KVD_ / 128, MROWS / 64), 128, 0, stream>>>(xh, Wvt, (void*)Vh, KVD_, D_);

  // RoPE on Q and K
  {
    int totq = B_ * N_ * HQ_ * 32;
    int totk = B_ * N_ * HKV_ * 32;
    rope_f16<<<(totq + TPB - 1) / TPB, TPB, 0, stream>>>(Qh, HQ_, totq);
    rope_f16<<<(totk + TPB - 1) / TPB, TPB, 0, stream>>>(Kh, HKV_, totk);
  }

  // V -> [b][hkv][hd][n]
  vtrans_f16<<<(B_ * KVD_ * N_ + TPB - 1) / TPB, TPB, 0, stream>>>(Vh, Vt);

  // flash attention: B*HQ*(N/16) waves, 4 waves/block
  fa_kernel<<<dim3(B_ * HQ_ * (N_ / 16) / 4), 128, 0, stream>>>(Qh, Kh, Vt, AOh);

  // output projection -> f32 result
  gemm_wmma<true><<<dim3(D_ / 128, MROWS / 64), 128, 0, stream>>>(AOh, Wot, (void*)out, D_, D_);
}
